// non_local_block_27487790695245
// MI455X (gfx1250) — matrix operand order repaired and checked by reading, not compiled, not run on hardware
//
#include <hip/hip_runtime.h>
#include <hip/hip_bf16.h>
#include <math.h>

// ---------- problem constants ----------
#define BATCH 8
#define C_CH  256          // C
#define CC    128          // c = C/2
#define N_PIX 4096         // H*W
#define M_PIX 1024         // pooled (H/2)*(W/2)

typedef _Float16 half8 __attribute__((ext_vector_type(8)));
typedef _Float16 v16h  __attribute__((ext_vector_type(16)));
typedef float    v8f   __attribute__((ext_vector_type(8)));
typedef float    f4v   __attribute__((ext_vector_type(4)));
typedef int      v4i   __attribute__((ext_vector_type(4)));

#define AS1 __attribute__((address_space(1)))
#define AS3 __attribute__((address_space(3)))

#if defined(__has_builtin)
#if __has_builtin(__builtin_amdgcn_global_load_async_to_lds_b128)
#define HAVE_ASYNC_LDS 1
#endif
#if __has_builtin(__builtin_amdgcn_s_wait_asynccnt)
#define HAVE_WAIT_ASYNC 1
#endif
#endif

// ---------- WMMA fragment helpers (CDNA5 16x16x32 f16, wave32) ----------
// A-matrix 16x32 f16: lane L holds row M=L&15; lanes<16 K-chunks {0..7,16..23},
// lanes>=16 K-chunks {8..15,24..31}. Two 16B loads.
__device__ inline v16h make_afrag(const _Float16* base, int stride) {
  const int lane = threadIdx.x & 31;
  const int row  = lane & 15;
  const int kb   = (lane & 16) ? 8 : 0;
  const _Float16* p = base + (size_t)row * stride + kb;
  union { v16h v; half8 h[2]; } u;
  u.h[0] = *(const half8*)(p);
  u.h[1] = *(const half8*)(p + 16);
  return u.v;
}

// B-matrix 32x16 f16: lane L holds column N=L&15; lanes<16 K {0..7,16..23},
// lanes>=16 K {8..15,24..31}.
__device__ inline v16h make_bfrag(const _Float16* base, int stride) {
  const int lane = threadIdx.x & 31;
  const int n = lane & 15;
  const int kb = (lane & 16) ? 8 : 0;
  v16h r;
#pragma unroll
  for (int i = 0; i < 8; ++i) {
    r[i]     = base[(size_t)(kb + i) * stride + n];
    r[8 + i] = base[(size_t)(16 + kb + i) * stride + n];
  }
  return r;
}

// B fragment sourced from f32 rows (converted on the fly).
__device__ inline v16h make_bfrag_f32(const float* base, int stride) {
  const int lane = threadIdx.x & 31;
  const int n = lane & 15;
  const int kb = (lane & 16) ? 8 : 0;
  v16h r;
#pragma unroll
  for (int i = 0; i < 8; ++i) {
    r[i]     = (_Float16)base[(size_t)(kb + i) * stride + n];
    r[8 + i] = (_Float16)base[(size_t)(16 + kb + i) * stride + n];
  }
  return r;
}

__device__ inline v8f wmma_f16(v16h a, v16h b, v8f c) {
  return __builtin_amdgcn_wmma_f32_16x16x32_f16(false, a, false, b, (short)0, c,
                                                false, false);
}

// ---------- Kernel W: convert weights to f16 ----------
__global__ __launch_bounds__(256) void cvt_weights_kernel(
    const float* __restrict__ thw, const float* __restrict__ phw,
    const float* __restrict__ gw,  const float* __restrict__ wz,
    _Float16* __restrict__ thwH, _Float16* __restrict__ phwH,
    _Float16* __restrict__ gwH,  _Float16* __restrict__ wzH) {
  int i = blockIdx.x * 256 + threadIdx.x;
  if (i < CC * C_CH) {
    thwH[i] = (_Float16)thw[i];
    phwH[i] = (_Float16)phw[i];
    gwH[i]  = (_Float16)gw[i];
    wzH[i]  = (_Float16)wz[i];
  }
}

// ---------- Kernel A: theta = conv1x1(x) -> thetaH [B][N][c] f16 ----------
// Computed as D^T = Wtheta(128x256) @ X(256x4096) per batch.
__global__ __launch_bounds__(256) void theta_kernel(
    const float* __restrict__ x, const _Float16* __restrict__ thwH,
    const float* __restrict__ theta_b, _Float16* __restrict__ thetaH) {
  const int bat = blockIdx.z;
  const int tid = threadIdx.x, lane = tid & 31, wave = tid >> 5;
  const int ostrip = wave & 3, nhalf = wave >> 2;
  const int obase = blockIdx.y * 64 + ostrip * 16;
  const int nbase = blockIdx.x * 128 + nhalf * 64;
  const int hoff = (lane & 16) ? 8 : 0;
  const float* xb = x + (size_t)bat * C_CH * N_PIX;

  v8f acc[4] = {};
  for (int k = 0; k < C_CH; k += 32) {
    v16h a = make_afrag(thwH + (size_t)obase * C_CH + k, C_CH);
#pragma unroll
    for (int t = 0; t < 4; ++t) {
      v16h bf = make_bfrag_f32(xb + (size_t)k * N_PIX + nbase + t * 16, N_PIX);
      acc[t] = wmma_f16(a, bf, acc[t]);
    }
  }
#pragma unroll
  for (int r = 0; r < 8; ++r) {
    int row = obase + r + hoff;
    float bias = theta_b[row];
#pragma unroll
    for (int t = 0; t < 4; ++t) {
      int n = nbase + t * 16 + (lane & 15);
      thetaH[(size_t)bat * N_PIX * CC + (size_t)n * CC + row] =
          (_Float16)(acc[t][r] + bias);
    }
  }
}

// ---------- Kernel B: maxpool + phi/g convs ----------
// phiH [B][c][M] (B-operand for scores), gH [B][M][c] (B-operand for P@G).
__global__ __launch_bounds__(256) void pool_phi_g_kernel(
    const float* __restrict__ x,
    const _Float16* __restrict__ phwH, const float* __restrict__ phi_b,
    const _Float16* __restrict__ gwH,  const float* __restrict__ g_b,
    _Float16* __restrict__ phiH, _Float16* __restrict__ gH) {
  __shared__ _Float16 xps[C_CH * 64];  // pooled tile [256 ch][64 m] f16, 32 KB
  const int bat = blockIdx.y;
  const int mt = blockIdx.x;           // 64-wide m tile (two pooled rows)
  const int tid = threadIdx.x, lane = tid & 31, wave = tid >> 5;
  const float* xb = x + (size_t)bat * C_CH * N_PIX;

  // 2x2 max pooling into LDS
  for (int i = 0; i < 64; ++i) {
    int e = tid + i * 256;
    int ch = e >> 6, m = e & 63;
    int pr = mt * 2 + (m >> 5), pc = m & 31;
    const float* p = xb + (size_t)ch * N_PIX + (size_t)(2 * pr) * 64 + 2 * pc;
    float v = fmaxf(fmaxf(p[0], p[1]), fmaxf(p[64], p[65]));
    xps[ch * 64 + m] = (_Float16)v;
  }
  __syncthreads();

  const int obase = wave * 16;  // 8 waves cover all 128 output channels
  const int hoff = (lane & 16) ? 8 : 0;
  v8f ap[4] = {}, ag[4] = {};
  for (int k = 0; k < C_CH; k += 32) {
    v16h aP = make_afrag(phwH + (size_t)obase * C_CH + k, C_CH);
    v16h aG = make_afrag(gwH + (size_t)obase * C_CH + k, C_CH);
#pragma unroll
    for (int t = 0; t < 4; ++t) {
      v16h bf = make_bfrag(xps + k * 64 + t * 16, 64);
      ap[t] = wmma_f16(aP, bf, ap[t]);
      ag[t] = wmma_f16(aG, bf, ag[t]);
    }
  }
#pragma unroll
  for (int r = 0; r < 8; ++r) {
    int row = obase + r + hoff;
    float bp = phi_b[row], bg = g_b[row];
#pragma unroll
    for (int t = 0; t < 4; ++t) {
      int m = mt * 64 + t * 16 + (lane & 15);
      phiH[(size_t)bat * CC * M_PIX + (size_t)row * M_PIX + m] =
          (_Float16)(ap[t][r] + bp);
      gH[(size_t)bat * M_PIX * CC + (size_t)m * CC + row] =
          (_Float16)(ag[t][r] + bg);
    }
  }
}

// ---------- Kernel C: fused attention (two-pass softmax) ----------
// Block = one (batch, 64-row n-tile). m2H [B][N][c] f16 (== m3 [B][c][N] flat).
__global__ __launch_bounds__(256) void attn_kernel(
    const _Float16* __restrict__ thetaH, const _Float16* __restrict__ phiH,
    const _Float16* __restrict__ gH, _Float16* __restrict__ m2H) {
  __shared__ _Float16 thS[64 * 136];  // theta tile, padded rows (17.4 KB)
  __shared__ float    Ssc[64 * 68];   // score chunk f32 (17.4 KB)
  __shared__ _Float16 Ps[64 * 72];    // softmaxed chunk f16 (9.2 KB)
  __shared__ float rowM[64];
  __shared__ float rowIL[64];
  __shared__ float redM[64 * 4];      // quarter-row max partials
  __shared__ float redS[64 * 4];      // quarter-row sumexp partials

  const int bat = blockIdx.y;
  const int nbase = blockIdx.x * 64;
  const int tid = threadIdx.x, lane = tid & 31, wave = tid >> 5;
  const int hoff = (lane & 16) ? 8 : 0;
  const int sstrip = wave & 3;   // 16-row strip
  const int shalf  = wave >> 2;  // 32-col half of 64-col S chunk
  const int chalf  = wave >> 2;  // 64-col half of c for O
  const int myrow = tid & 63;    // row owned for softmax phases
  const int myq   = tid >> 6;    // quarter (16 cols) of that row

  // ---- stage theta tile (64 rows x 128 k) into LDS ----
  const _Float16* thSrc = thetaH + (size_t)bat * N_PIX * CC + (size_t)nbase * CC;
#if HAVE_ASYNC_LDS
  {
    _Float16* srcNC = const_cast<_Float16*>(thSrc);
#pragma unroll
    for (int i = 0; i < 4; ++i) {
      int idx = tid + i * 256;     // 16B units (1024 total)
      int r = idx >> 4, kc = idx & 15;
      __builtin_amdgcn_global_load_async_to_lds_b128(
          (AS1 v4i*)(srcNC + (size_t)r * 128 + kc * 8),
          (AS3 v4i*)(thS + r * 136 + kc * 8), 0, 0);
    }
  }
#if HAVE_WAIT_ASYNC
  __builtin_amdgcn_s_wait_asynccnt(0);
#else
  asm volatile("s_wait_asynccnt 0x0" ::: "memory");
#endif
#else
#pragma unroll
  for (int i = 0; i < 4; ++i) {
    int idx = tid + i * 256;       // half8 units (1024 total)
    int r = idx >> 4, kc = idx & 15;
    *(half8*)(thS + r * 136 + kc * 8) = *(const half8*)(thSrc + r * 128 + kc * 8);
  }
#endif
  __syncthreads();

  const _Float16* phiB = phiH + (size_t)bat * CC * M_PIX;

  auto computeS = [&](int mchunk) {
    v8f a0 = {}, a1 = {};
    const int mcol = mchunk * 64 + shalf * 32;
#pragma unroll
    for (int k = 0; k < CC; k += 32) {
      v16h a = make_afrag(thS + sstrip * 16 * 136 + k, 136);
      v16h b0 = make_bfrag(phiB + (size_t)k * M_PIX + mcol, M_PIX);
      v16h b1 = make_bfrag(phiB + (size_t)k * M_PIX + mcol + 16, M_PIX);
      a0 = wmma_f16(a, b0, a0);
      a1 = wmma_f16(a, b1, a1);
    }
#pragma unroll
    for (int r = 0; r < 8; ++r) {
      int row = sstrip * 16 + r + hoff;
      Ssc[row * 68 + shalf * 32 + (lane & 15)] = a0[r];
      Ssc[row * 68 + shalf * 32 + 16 + (lane & 15)] = a1[r];
    }
  };

  // ---- pass 1: streaming row max / sum-of-exp over all 16 chunks ----
  // All 256 threads reduce a 16-wide quarter-row to (max, sumexp) partials;
  // the 64 row-owner threads merge 4 partials into running stats.
  float runM = -3.0e38f, runL = 0.0f;
  for (int mc = 0; mc < 16; ++mc) {
    computeS(mc);
    __syncthreads();
    {
      const float* src = &Ssc[myrow * 68 + myq * 16];
      float qm = -3.0e38f;
      for (int j = 0; j < 16; ++j) qm = fmaxf(qm, src[j]);
      float qs = 0.0f;
      for (int j = 0; j < 16; ++j) qs += __expf(src[j] - qm);
      redM[myrow * 4 + myq] = qm;
      redS[myrow * 4 + myq] = qs;
    }
    __syncthreads();
    if (tid < 64) {
      float M = runM, L = runL;
#pragma unroll
      for (int q = 0; q < 4; ++q) {
        float mq = redM[tid * 4 + q], lq = redS[tid * 4 + q];
        float nm = fmaxf(M, mq);
        L = L * __expf(M - nm) + lq * __expf(mq - nm);
        M = nm;
      }
      runM = M; runL = L;
    }
    __syncthreads();
  }
  if (tid < 64) { rowM[tid] = runM; rowIL[tid] = 1.0f / runL; }
  __syncthreads();

  // ---- pass 2: recompute S, P = exp(S-m)/l, O += P @ G ----
  const _Float16* gB = gH + (size_t)bat * M_PIX * CC;
  v8f o[4] = {};
  for (int mc = 0; mc < 16; ++mc) {
    // prefetch next chunk of G while computing S (global_prefetch_b8)
    if (mc + 1 < 16)
      __builtin_prefetch(gB + (size_t)((mc + 1) * 64) * CC + (size_t)tid * 32,
                         0, 0);
    computeS(mc);
    __syncthreads();
    {
      float m = rowM[myrow], il = rowIL[myrow];
      const float* src = &Ssc[myrow * 68 + myq * 16];
      _Float16* dst = &Ps[myrow * 72 + myq * 16];
      for (int j = 0; j < 16; ++j)
        dst[j] = (_Float16)(__expf(src[j] - m) * il);
    }
    __syncthreads();
#pragma unroll
    for (int ks = 0; ks < 2; ++ks) {
      v16h a = make_afrag(Ps + sstrip * 16 * 72 + ks * 32, 72);
#pragma unroll
      for (int t = 0; t < 4; ++t) {
        v16h bf = make_bfrag(gB + (size_t)(mc * 64 + ks * 32) * CC +
                                 chalf * 64 + t * 16, CC);
        o[t] = wmma_f16(a, bf, o[t]);
      }
    }
    __syncthreads();
  }

#pragma unroll
  for (int t = 0; t < 4; ++t)
#pragma unroll
    for (int r = 0; r < 8; ++r) {
      int row = sstrip * 16 + r + hoff;
      int ch = chalf * 64 + t * 16 + (lane & 15);
      m2H[(size_t)bat * N_PIX * CC + (size_t)(nbase + row) * CC + ch] =
          (_Float16)o[t][r];
    }
}

// ---------- Kernel D: out = Wz @ m3 + bias + x ----------
__global__ __launch_bounds__(256) void out_kernel(
    const _Float16* __restrict__ m2H, const _Float16* __restrict__ wzH,
    const float* __restrict__ wz_b, const float* __restrict__ x,
    float* __restrict__ out) {
  const int bat = blockIdx.z;
  const int tid = threadIdx.x, lane = tid & 31, wave = tid >> 5;
  const int ostrip = wave & 3, phalf = wave >> 2;
  const int obase = blockIdx.y * 64 + ostrip * 16;
  const int pbase = blockIdx.x * 128 + phalf * 64;
  const int hoff = (lane & 16) ? 8 : 0;
  // raw reinterpretation: m2 flat [N][c] == m3 [c][N] (rows contiguous in p)
  const _Float16* m3 = m2H + (size_t)bat * CC * N_PIX;

  v8f acc[4] = {};
  for (int k = 0; k < CC; k += 32) {
    v16h a = make_afrag(wzH + (size_t)obase * CC + k, CC);
#pragma unroll
    for (int t = 0; t < 4; ++t) {
      v16h bf = make_bfrag(m3 + (size_t)k * N_PIX + pbase + t * 16, N_PIX);
      acc[t] = wmma_f16(a, bf, acc[t]);
    }
  }
#pragma unroll
  for (int r = 0; r < 8; ++r) {
    int row = obase + r + hoff;
    float bias = wz_b[row];
#pragma unroll
    for (int t = 0; t < 4; ++t) {
      int p = pbase + t * 16 + (lane & 15);
      size_t idx = (size_t)bat * C_CH * N_PIX + (size_t)row * N_PIX + p;
      out[idx] = acc[t][r] + bias + x[idx];
    }
  }
}

// ---------- host launcher ----------
extern "C" void kernel_launch(void* const* d_in, const int* in_sizes, int n_in,
                              void* d_out, int out_size, void* d_ws, size_t ws_size,
                              hipStream_t stream) {
  const float* x       = (const float*)d_in[0];
  const float* theta_w = (const float*)d_in[1];
  const float* theta_b = (const float*)d_in[2];
  const float* phi_w   = (const float*)d_in[3];
  const float* phi_b   = (const float*)d_in[4];
  const float* g_w     = (const float*)d_in[5];
  const float* g_b     = (const float*)d_in[6];
  const float* wz_w    = (const float*)d_in[7];
  const float* wz_b    = (const float*)d_in[8];
  float* out = (float*)d_out;

  char* ws = (char*)d_ws;
  _Float16* thetaH = (_Float16*)(ws);                         // 8 MB
  _Float16* phiH   = (_Float16*)(ws + (8u << 20));            // 2 MB
  _Float16* gH     = (_Float16*)(ws + (10u << 20));           // 2 MB
  _Float16* m2H    = (_Float16*)(ws + (12u << 20));           // 8 MB
  _Float16* thwH   = (_Float16*)(ws + (20u << 20));
  _Float16* phwH   = thwH + CC * C_CH;
  _Float16* gwH    = phwH + CC * C_CH;
  _Float16* wzH    = gwH + CC * C_CH;

  cvt_weights_kernel<<<dim3(128), 256, 0, stream>>>(
      theta_w, phi_w, g_w, wz_w, thwH, phwH, gwH, wzH);

  theta_kernel<<<dim3(32, 2, BATCH), 256, 0, stream>>>(
      x, thwH, theta_b, thetaH);

  pool_phi_g_kernel<<<dim3(16, BATCH), 256, 0, stream>>>(
      x, phwH, phi_b, gwH, g_b, phiH, gH);

  attn_kernel<<<dim3(64, BATCH), 256, 0, stream>>>(
      thetaH, phiH, gH, m2H);

  out_kernel<<<dim3(32, 4, BATCH), 256, 0, stream>>>(
      m2H, wzH, wz_b, x, out);
}
